// GAT_37718402794124
// MI455X (gfx1250) — compile-verified, matrix operand order checked by reading, not run on hardware
//
#include <hip/hip_runtime.h>
#include <hip/hip_bf16.h>

#define NODES 50000
#define EDGES 500000
#define NGRAPH 64
#define HEADS 2
#define DHID 64
#define FDIM 128          // HEADS*DHID
#define FIN 4
#define NCLS 5
#define NEG_SLOPE 0.2f
#define NBLK_SCAN 196     // ceil(NODES/256)

typedef __attribute__((ext_vector_type(16))) __bf16 v16bf;
typedef __attribute__((ext_vector_type(8)))  __bf16 v8bf;
typedef __attribute__((ext_vector_type(4)))  __bf16 v4bf;
typedef __attribute__((ext_vector_type(8)))  float  v8f;

// ------------------------------------------------- layer 1 GEMM (K=4, scalar)
__global__ void __launch_bounds__(FDIM)
gemm_fin4(const float* __restrict__ x, const float* __restrict__ W,
          float* __restrict__ h) {
  const int n = blockIdx.x;
  const int j = threadIdx.x;
  const float x0 = x[n * FIN + 0], x1 = x[n * FIN + 1];
  const float x2 = x[n * FIN + 2], x3 = x[n * FIN + 3];
  h[(size_t)n * FDIM + j] =
      x0 * W[0 * FDIM + j] + x1 * W[1 * FDIM + j] +
      x2 * W[2 * FDIM + j] + x3 * W[3 * FDIM + j];
}

// ------------------------------------------- W (128x128 f32) -> col-major bf16
__global__ void __launch_bounds__(256)
convW(const float* __restrict__ W, __bf16* __restrict__ Wt) {
  const int idx = blockIdx.x * 256 + threadIdx.x;    // 16384 elems, exact
  const int k = idx >> 7, j = idx & 127;
  Wt[j * FDIM + k] = (__bf16)W[idx];                 // transpose to [N][K]
}

// ------------------------------------------------ WMMA GEMM: [N,128]x[128,128]
// Block = 256 threads = 8 wave32; block computes 16 rows x 128 cols.
// Wave w computes the 16x16 tile at columns [16w,16w+16), K-loop of 4 x 32.
__global__ void __launch_bounds__(256)
gemm_bf16_wmma(const __bf16* __restrict__ A, const __bf16* __restrict__ Bt,
               float* __restrict__ C) {
  __shared__ __align__(16) __bf16 As[16 * FDIM];     // 16x128 bf16 tile = 4 KB
  const int m0 = blockIdx.x * 16;

  const unsigned* Ag   = (const unsigned*)(A + (size_t)m0 * FDIM);
  unsigned*       As32 = (unsigned*)As;
#pragma unroll
  for (int i = 0; i < 4; ++i)
    As32[threadIdx.x + i * 256] = Ag[threadIdx.x + i * 256];
  __syncthreads();

  const int lane = threadIdx.x & 31;
  const int wave = threadIdx.x >> 5;
  const int row  = lane & 15;        // A-row / B-col within tile
  const int hi   = lane >> 4;        // which K-half this lane carries
  const int col  = (wave << 4) + row;

  v8f c = {};
#pragma unroll
  for (int kb = 0; kb < 4; ++kb) {
    const __bf16* ap = As + row * FDIM + kb * 32 + hi * 8;
    v8bf alo = *(const v8bf*)(ap);
    v8bf ahi = *(const v8bf*)(ap + 16);
    v16bf a  = __builtin_shufflevector(alo, ahi,
                 0,1,2,3,4,5,6,7,8,9,10,11,12,13,14,15);
    const __bf16* bp = Bt + (size_t)col * FDIM + kb * 32 + hi * 8;
    v8bf blo = *(const v8bf*)(bp);
    v8bf bhi = *(const v8bf*)(bp + 16);
    v16bf b  = __builtin_shufflevector(blo, bhi,
                 0,1,2,3,4,5,6,7,8,9,10,11,12,13,14,15);
    c = __builtin_amdgcn_wmma_f32_16x16x32_bf16(
            false, a, false, b, (short)0, c, false, false);
  }
#pragma unroll
  for (int r = 0; r < 8; ++r) {
    const int m = m0 + r + hi * 8;
    C[(size_t)m * FDIM + col] = c[r];
  }
}

// ------------------------------------------------ per-node attention logits
__global__ void __launch_bounds__(256)
alpha_kernel(const float* __restrict__ h, const float* __restrict__ asv,
             const float* __restrict__ adv, float* __restrict__ als,
             float* __restrict__ ald) {
  const int id = blockIdx.x * 256 + threadIdx.x;
  if (id >= NODES * HEADS) return;
  const int n = id >> 1, hh = id & 1;
  const float4* hp = (const float4*)(h + (size_t)n * FDIM + hh * DHID);
  const float4* sp = (const float4*)(asv + hh * DHID);
  const float4* dp = (const float4*)(adv + hh * DHID);
  float s1 = 0.f, s2 = 0.f;
#pragma unroll
  for (int i = 0; i < 16; ++i) {
    const float4 hv = hp[i], a1 = sp[i], a2 = dp[i];
    s1 += hv.x * a1.x + hv.y * a1.y + hv.z * a1.z + hv.w * a1.w;
    s2 += hv.x * a2.x + hv.y * a2.y + hv.z * a2.z + hv.w * a2.w;
  }
  als[id] = s1;
  ald[id] = s2;
}

// =========================== CSR construction (once per call; graph is static)
__global__ void __launch_bounds__(256)
init_deg(int* __restrict__ deg) {
  const int i = blockIdx.x * 256 + threadIdx.x;
  if (i < NODES) deg[i] = 0;
}

__global__ void __launch_bounds__(256)
hist_deg(const int* __restrict__ dstv, int* __restrict__ deg) {
  const int e = blockIdx.x * 256 + threadIdx.x;
  if (e < EDGES) atomicAdd(&deg[dstv[e]], 1);
}

// block-local Hillis-Steele exclusive scan; emits per-block totals
__global__ void __launch_bounds__(256)
scan_local(const int* __restrict__ deg, int* __restrict__ partial,
           int* __restrict__ bsum) {
  __shared__ int lds[256];
  const int gid = blockIdx.x * 256 + threadIdx.x;
  const int v = (gid < NODES) ? deg[gid] : 0;
  lds[threadIdx.x] = v;
  __syncthreads();
#pragma unroll
  for (int off = 1; off < 256; off <<= 1) {
    const int t = (threadIdx.x >= off) ? lds[threadIdx.x - off] : 0;
    __syncthreads();
    lds[threadIdx.x] += t;
    __syncthreads();
  }
  if (gid < NODES) partial[gid] = lds[threadIdx.x] - v;  // exclusive
  if (threadIdx.x == 255) bsum[blockIdx.x] = lds[255];
}

__global__ void __launch_bounds__(256)
scan_bsum(const int* __restrict__ bsum, int* __restrict__ boff) {
  __shared__ int lds[256];
  const int v = (threadIdx.x < NBLK_SCAN) ? bsum[threadIdx.x] : 0;
  lds[threadIdx.x] = v;
  __syncthreads();
#pragma unroll
  for (int off = 1; off < 256; off <<= 1) {
    const int t = (threadIdx.x >= off) ? lds[threadIdx.x - off] : 0;
    __syncthreads();
    lds[threadIdx.x] += t;
    __syncthreads();
  }
  if (threadIdx.x < NBLK_SCAN) boff[threadIdx.x] = lds[threadIdx.x] - v;
}

__global__ void __launch_bounds__(256)
scan_finalize(const int* __restrict__ partial, const int* __restrict__ boff,
              int* __restrict__ starts, int* __restrict__ cursor) {
  const int gid = blockIdx.x * 256 + threadIdx.x;
  if (gid < NODES) {
    const int s = partial[gid] + boff[blockIdx.x];
    starts[gid] = s;
    cursor[gid] = s;
  }
  if (gid == 0) starts[NODES] = EDGES;
}

__global__ void __launch_bounds__(256)
build_csr(const int* __restrict__ src, const int* __restrict__ dstv,
          int* __restrict__ cursor, int* __restrict__ csr_src) {
  const int e = blockIdx.x * 256 + threadIdx.x;
  if (e >= EDGES) return;
  const int p = atomicAdd(&cursor[dstv[e]], 1);
  csr_src[p] = src[e];     // store source node id directly
}

// ============= fused per-dst softmax + aggregation + bias + ReLU + bf16 cast
// One wave32 per destination node: shfl-reduced max & exp-sum over in-edges,
// then each lane owns one float4 chunk of the 128-wide feature. Zero fp atomics;
// h (25.6 MB) is L2-resident so the per-edge row gathers hit L2.
__global__ void __launch_bounds__(256)
gat_aggregate(const int* __restrict__ starts, const int* __restrict__ csr_src,
              const float* __restrict__ als, const float* __restrict__ ald,
              const float* __restrict__ h, const float* __restrict__ bias,
              float* __restrict__ xf, __bf16* __restrict__ xbf) {
  const int d    = (blockIdx.x * 256 + threadIdx.x) >> 5;  // grid exact
  const int lane = threadIdx.x & 31;
  const int s0   = starts[d];
  const int deg  = starts[d + 1] - s0;
  const float ad0 = ald[d * 2 + 0], ad1 = ald[d * 2 + 1];

  // pass 1: segment max (leaky-relu'd logits)
  float m0 = -__builtin_inff(), m1 = -__builtin_inff();
  for (int i = lane; i < deg; i += 32) {
    const int s = csr_src[s0 + i];
    float l0 = als[s * 2 + 0] + ad0; l0 = l0 > 0.f ? l0 : NEG_SLOPE * l0;
    float l1 = als[s * 2 + 1] + ad1; l1 = l1 > 0.f ? l1 : NEG_SLOPE * l1;
    m0 = fmaxf(m0, l0); m1 = fmaxf(m1, l1);
  }
#pragma unroll
  for (int o = 16; o; o >>= 1) {
    m0 = fmaxf(m0, __shfl_xor(m0, o, 32));
    m1 = fmaxf(m1, __shfl_xor(m1, o, 32));
  }

  // pass 2: exp-sum
  float sum0 = 0.f, sum1 = 0.f;
  for (int i = lane; i < deg; i += 32) {
    const int s = csr_src[s0 + i];
    float l0 = als[s * 2 + 0] + ad0; l0 = l0 > 0.f ? l0 : NEG_SLOPE * l0;
    float l1 = als[s * 2 + 1] + ad1; l1 = l1 > 0.f ? l1 : NEG_SLOPE * l1;
    sum0 += __expf(l0 - m0); sum1 += __expf(l1 - m1);
  }
#pragma unroll
  for (int o = 16; o; o >>= 1) {
    sum0 += __shfl_xor(sum0, o, 32);
    sum1 += __shfl_xor(sum1, o, 32);
  }

  // pass 3: weighted feature accumulation; lane owns float4 chunk `lane`
  const int   hh   = lane >> 4;
  const float mh   = hh ? m1 : m0;
  const float invh = 1.f / ((hh ? sum1 : sum0) + 1e-16f);
  const float adh  = hh ? ad1 : ad0;
  float4 a4 = *(const float4*)(bias + lane * 4);
  for (int i = 0; i < deg; ++i) {
    const int s = csr_src[s0 + i];                 // wave-uniform broadcast
    float l = als[s * 2 + hh] + adh; l = l > 0.f ? l : NEG_SLOPE * l;
    const float alpha = __expf(l - mh) * invh;
    const float4 hv = *(const float4*)(h + (size_t)s * FDIM + lane * 4);
    a4.x += alpha * hv.x; a4.y += alpha * hv.y;
    a4.z += alpha * hv.z; a4.w += alpha * hv.w;
  }
  // ReLU + f32 out + bf16 out for next layer's WMMA
  a4.x = fmaxf(a4.x, 0.f); a4.y = fmaxf(a4.y, 0.f);
  a4.z = fmaxf(a4.z, 0.f); a4.w = fmaxf(a4.w, 0.f);
  *(float4*)(xf + (size_t)d * FDIM + lane * 4) = a4;
  v4bf b4 = { (__bf16)a4.x, (__bf16)a4.y, (__bf16)a4.z, (__bf16)a4.w };
  *(v4bf*)(xbf + (size_t)d * FDIM + lane * 4) = b4;
}

// ------------------------------------------------------------- pooling
__global__ void __launch_bounds__(256)
init_pool(float* __restrict__ pooled, float* __restrict__ counts) {
  const int idx = blockIdx.x * 256 + threadIdx.x;
  if (idx < NGRAPH * FDIM) pooled[idx] = 0.f;
  if (idx < NGRAPH)        counts[idx] = 0.f;
}

// batch is sorted: accumulate a 256-node strip per column, flush on boundary
__global__ void __launch_bounds__(FDIM)
pool_kernel(const float* __restrict__ h, const int* __restrict__ batch,
            float* __restrict__ pooled, float* __restrict__ counts) {
  const int n0  = blockIdx.x * 256;
  const int j   = threadIdx.x;
  const int end = (n0 + 256 < NODES) ? n0 + 256 : NODES;
  float av = 0.f;
  int   cg = batch[n0];
  for (int n = n0; n < end; ++n) {
    const int g = batch[n];
    if (g != cg) { atomicAdd(&pooled[cg * FDIM + j], av); av = 0.f; cg = g; }
    av += h[(size_t)n * FDIM + j];
  }
  atomicAdd(&pooled[cg * FDIM + j], av);
  if (j == 0) {
    float cnt = 0.f;
    int   g0  = batch[n0];
    for (int n = n0; n < end; ++n) {
      const int g = batch[n];
      if (g != g0) { atomicAdd(&counts[g0], cnt); cnt = 0.f; g0 = g; }
      cnt += 1.f;
    }
    atomicAdd(&counts[g0], cnt);
  }
}

// ------------------------------------------------------- classifier head
__global__ void __launch_bounds__(64)
head_kernel(const float* __restrict__ pooled, const float* __restrict__ counts,
            const float* __restrict__ lw, const float* __restrict__ lb,
            float* __restrict__ out) {
  const int g = threadIdx.x;
  if (g >= NGRAPH) return;
  float cnt = counts[g];
  cnt = cnt > 1.f ? cnt : 1.f;
  const float inv = 1.f / cnt;
  float z[NCLS];
#pragma unroll
  for (int c = 0; c < NCLS; ++c) z[c] = lb[c];
  for (int j = 0; j < FDIM; ++j) {
    const float m = pooled[g * FDIM + j] * inv;
#pragma unroll
    for (int c = 0; c < NCLS; ++c) z[c] += m * lw[j * NCLS + c];
  }
#pragma unroll
  for (int c = 0; c < NCLS; ++c)
    out[g * NCLS + c] = 1.f / (1.f + __expf(-z[c]));
}

// ----------------------------------------------------------------- launcher
extern "C" void kernel_launch(void* const* d_in, const int* in_sizes, int n_in,
                              void* d_out, int out_size, void* d_ws, size_t ws_size,
                              hipStream_t stream) {
  (void)in_sizes; (void)n_in; (void)out_size; (void)ws_size;

  char* ws = (char*)d_ws;
  auto take = [&](size_t bytes) -> char* {
    char* p = ws;
    ws += (bytes + 255) & ~(size_t)255;
    return p;
  };
  __bf16* xbf     = (__bf16*)take((size_t)NODES * FDIM * sizeof(__bf16));
  float*  h       = (float*) take((size_t)NODES * FDIM * sizeof(float));
  float*  xf      = (float*) take((size_t)NODES * FDIM * sizeof(float));
  float*  als     = (float*) take((size_t)NODES * HEADS * sizeof(float));
  float*  ald     = (float*) take((size_t)NODES * HEADS * sizeof(float));
  __bf16* wbf     = (__bf16*)take((size_t)FDIM * FDIM * sizeof(__bf16));
  int*    deg     = (int*)   take((size_t)NODES * sizeof(int));
  int*    partial = (int*)   take((size_t)NODES * sizeof(int));
  int*    bsum    = (int*)   take(256 * sizeof(int));
  int*    boff    = (int*)   take(256 * sizeof(int));
  int*    starts  = (int*)   take((size_t)(NODES + 1) * sizeof(int));
  int*    cursor  = (int*)   take((size_t)NODES * sizeof(int));
  int*    csr_src = (int*)   take((size_t)EDGES * sizeof(int));
  float*  pooled  = (float*) take((size_t)NGRAPH * FDIM * sizeof(float));
  float*  counts  = (float*) take((size_t)NGRAPH * sizeof(float));

  const float* x     = (const float*)d_in[0];
  const int*   ei    = (const int*)  d_in[1];
  const int*   src   = ei;
  const int*   dstv  = ei + EDGES;
  const int*   batch = (const int*)  d_in[2];
  const float* Wl[4], *asl[4], *adl[4], *bl[4];
  for (int l = 0; l < 4; ++l) {
    Wl[l]  = (const float*)d_in[3 + 4 * l];
    asl[l] = (const float*)d_in[4 + 4 * l];
    adl[l] = (const float*)d_in[5 + 4 * l];
    bl[l]  = (const float*)d_in[6 + 4 * l];
  }
  const float* lw  = (const float*)d_in[19];
  const float* lb  = (const float*)d_in[20];
  float*       out = (float*)d_out;

  const int gNode = NBLK_SCAN;                 // ceil(NODES/256)
  const int gEdge = (EDGES + 255) / 256;
  const int gAl   = (NODES * HEADS + 255) / 256;
  const int gAgg  = (NODES * 32) / 256;        // 6250, exact

  // ---- CSR by destination (graph is static across all 4 layers)
  init_deg     <<<gNode, 256, 0, stream>>>(deg);
  hist_deg     <<<gEdge, 256, 0, stream>>>(dstv, deg);
  scan_local   <<<gNode, 256, 0, stream>>>(deg, partial, bsum);
  scan_bsum    <<<1,     256, 0, stream>>>(bsum, boff);
  scan_finalize<<<gNode, 256, 0, stream>>>(partial, boff, starts, cursor);
  build_csr    <<<gEdge, 256, 0, stream>>>(src, dstv, cursor, csr_src);

  // ---- 4 GAT layers
  for (int l = 0; l < 4; ++l) {
    if (l == 0) {
      gemm_fin4<<<NODES, FDIM, 0, stream>>>(x, Wl[0], h);
    } else {
      convW<<<(FDIM * FDIM) / 256, 256, 0, stream>>>(Wl[l], wbf);
      gemm_bf16_wmma<<<NODES / 16, 256, 0, stream>>>(xbf, wbf, h);
    }
    alpha_kernel <<<gAl,  256, 0, stream>>>(h, asl[l], adl[l], als, ald);
    gat_aggregate<<<gAgg, 256, 0, stream>>>(starts, csr_src, als, ald, h,
                                            bl[l], xf, xbf);
  }

  // ---- mean pool + sigmoid head
  init_pool  <<<(NGRAPH * FDIM + 255) / 256, 256, 0, stream>>>(pooled, counts);
  pool_kernel<<<gNode, FDIM, 0, stream>>>(xf, batch, pooled, counts);
  head_kernel<<<1, 64, 0, stream>>>(pooled, counts, lw, lb, out);
}
